// SelfAttention_63737314672692
// MI455X (gfx1250) — matrix operand order repaired and checked by reading, not compiled, not run on hardware
//
#include <hip/hip_runtime.h>
#include <hip/hip_bf16.h>
#include <stdint.h>

#define EMB   1024
#define HD    128
#define TLEN  2048
#define BATCH 8
#define NROWS (BATCH * TLEN)   // 16384

typedef __attribute__((ext_vector_type(16))) __bf16       v16bf;
typedef __attribute__((ext_vector_type(8)))  __bf16       v8bf;
typedef __attribute__((ext_vector_type(8)))  float        v8f;
typedef __attribute__((ext_vector_type(4)))  unsigned int u32x4;
typedef __attribute__((ext_vector_type(8)))  int          i32x8;
typedef __attribute__((ext_vector_type(4)))  int          i32x4;

// ---------------- WMMA helpers ----------------

__device__ __forceinline__ v8f wmma_bf16(v16bf a, v16bf b, v8f c) {
  // D = A(16x32 bf16) * B(32x16 bf16) + C(16x16 f32)
  return __builtin_amdgcn_wmma_f32_16x16x32_bf16(false, a, false, b,
                                                 (short)0, c, false, false);
}

// A-matrix fragment (16x32, 16-bit): lanes 0-15 -> K {0..7,16..23},
// lanes 16-31 -> K {8..15,24..31}. Row-major source tile, row stride `ld`.
__device__ __forceinline__ v16bf frag_a(const __bf16* tile, int row, int ld) {
  const unsigned lane = threadIdx.x & 31u;
  const __bf16* p = tile + (size_t)row * ld + ((lane & 16u) ? 8 : 0);
  v8bf lo = *(const v8bf*)p;
  v8bf hi = *(const v8bf*)(p + 16);
  v16bf f;
#pragma unroll
  for (int i = 0; i < 8; ++i) { f[i] = lo[i]; f[i + 8] = hi[i]; }
  return f;
}

// B-matrix fragment (32x16, 16-bit): lanes 0-15 -> K {0..7,16..23} (col = lane),
// lanes 16-31 -> K {8..15,24..31} (col = lane-16). `row` selects the source row that
// supplies column (lane&15) of B (i.e. B is Source^T), row stride `ld`.
__device__ __forceinline__ v16bf frag_b(const __bf16* tile, int row, int ld) {
  const unsigned lane = threadIdx.x & 31u;
  const __bf16* p = tile + (size_t)row * ld + ((lane & 16u) ? 8 : 0);
  v8bf lo = *(const v8bf*)p;
  v8bf hi = *(const v8bf*)(p + 16);
  v16bf f;
#pragma unroll
  for (int i = 0; i < 8; ++i) { f[i] = lo[i]; f[i + 8] = hi[i]; }
  return f;
}

// ---------------- TDM: 2-D bf16 tile -> LDS ----------------

__device__ __forceinline__ void tdm_load_2d_bf16(unsigned lds_off, const void* gptr,
                                                 unsigned rows, unsigned cols,
                                                 unsigned stride_elems) {
  const unsigned long long ga = (unsigned long long)(uintptr_t)gptr;
  u32x4 g0;
  g0[0] = 1u;                                        // count=1 valid user D#
  g0[1] = lds_off;                                   // lds_addr (bytes)
  g0[2] = (unsigned)(ga & 0xFFFFFFFFull);            // global_addr[31:0]
  g0[3] = (unsigned)((ga >> 32) & 0x1FFFFFFull)      // global_addr[56:32]
          | (2u << 30);                              // type = 2 ("image")
  i32x8 g1;
  g1[0] = (int)(1u << 16);                           // wg_mask=0, data_size=1 (2B)
  g1[1] = (int)((cols & 0xFFFFu) << 16);             // tensor_dim0[15:0] @63:48
  g1[2] = (int)(((cols >> 16) & 0xFFFFu) |           // tensor_dim0[31:16]
                ((rows & 0xFFFFu) << 16));           // tensor_dim1[15:0]
  g1[3] = (int)(((rows >> 16) & 0xFFFFu) |           // tensor_dim1[31:16]
                ((cols & 0xFFFFu) << 16));           // tile_dim0 = cols
  g1[4] = (int)(rows & 0xFFFFu);                     // tile_dim1 = rows, tile_dim2=0
  g1[5] = (int)stride_elems;                         // tensor_dim0_stride[31:0]
  g1[6] = 0;                                         // stride hi, dim1_stride lo
  g1[7] = 0;
  i32x4 g2 = {0, 0, 0, 0};
  i32x4 g3 = {0, 0, 0, 0};
  i32x8 g4 = {0, 0, 0, 0, 0, 0, 0, 0};               // extra operand (6-arg form)
  __builtin_amdgcn_tensor_load_to_lds(g0, g1, g2, g3, g4, 0);
}

// ---------------- Kernel 1: QKV projection (x @ W^T, bf16 out) ----------------

__global__ __launch_bounds__(256) void qkv_proj_kernel(
    const float* __restrict__ x, const float* __restrict__ Wq,
    const float* __restrict__ Wk, const float* __restrict__ Wv,
    unsigned short* __restrict__ qkv) {
  __shared__ __bf16 xs[128][32];
  __shared__ __bf16 wsh[128][32];

  const int m0   = blockIdx.x * 128;
  const int sel  = blockIdx.y;                 // 0=q, 1=k, 2=v
  const float* W = (sel == 0) ? Wq : ((sel == 1) ? Wk : Wv);
  __bf16* out = reinterpret_cast<__bf16*>(qkv) + (size_t)sel * NROWS * HD;

  const int tid = threadIdx.x;
  const int wave = tid >> 5, lane = tid & 31;

  v8f acc[8];
#pragma unroll
  for (int j = 0; j < 8; ++j) acc[j] = v8f{};

  for (int k0 = 0; k0 < EMB; k0 += 32) {
    // cooperative fp32 -> bf16 staging: 2 threads per row, 16 elems each
    {
      const int row = tid >> 1;
      const int c   = (tid & 1) * 16;
      const float* gx = x + (size_t)(m0 + row) * EMB + k0 + c;
      const float* gw = W + (size_t)row * EMB + k0 + c;
#pragma unroll
      for (int i = 0; i < 16; ++i) xs[row][c + i] = (__bf16)gx[i];
#pragma unroll
      for (int i = 0; i < 16; ++i) wsh[row][c + i] = (__bf16)gw[i];
    }
    __syncthreads();

    const v16bf a = frag_a(&xs[0][0], wave * 16 + (lane & 15), 32);
#pragma unroll
    for (int j = 0; j < 8; ++j) {
      const v16bf b = frag_b(&wsh[0][0], j * 16 + (lane & 15), 32);
      acc[j] = wmma_bf16(a, b, acc[j]);
    }
    __syncthreads();
  }

  const int hi = lane >> 4, col = lane & 15;
#pragma unroll
  for (int j = 0; j < 8; ++j)
#pragma unroll
    for (int r = 0; r < 8; ++r) {
      const int row = m0 + wave * 16 + r + 8 * hi;
      out[(size_t)row * HD + j * 16 + col] = (__bf16)acc[j][r];
    }
}

// ---------------- Kernel 2: causal flash attention ----------------

__global__ __launch_bounds__(256) void attn_kernel(
    const unsigned short* __restrict__ qkv, float* __restrict__ outp) {
  __shared__ __bf16 ksh[2][32][HD];     // K tile, row-major [key][dim]
  __shared__ __bf16 vT[2][HD][32];      // V tile transposed [dim][key]
  __shared__ __bf16 psh[8][16][32];     // per-wave P staging (C-layout -> A-layout)

  const int tid = threadIdx.x, wave = tid >> 5, lane = tid & 31;
  const int b  = blockIdx.y;
  const int m0 = blockIdx.x * 128;
  const __bf16* q  = reinterpret_cast<const __bf16*>(qkv);
  const __bf16* kk = q + (size_t)NROWS * HD;
  const __bf16* vv = kk + (size_t)NROWS * HD;
  const size_t rowbase = (size_t)b * TLEN;

  // q fragments live in registers for the whole kernel (4 chunks of K=32)
  v16bf qf[4];
#pragma unroll
  for (int c = 0; c < 4; ++c)
    qf[c] = frag_a(q + rowbase * HD + c * 32, m0 + wave * 16 + (lane & 15), HD);

  v8f oacc[8];
#pragma unroll
  for (int j = 0; j < 8; ++j) oacc[j] = v8f{};
  float mst[8], lst[8];
#pragma unroll
  for (int r = 0; r < 8; ++r) { mst[r] = -__builtin_inff(); lst[r] = 0.f; }

  const int   hi    = lane >> 4;
  const int   nblk  = (m0 + 128) / 32;
  const float scale = 0.08838834764831845f;   // 128^-0.5

  const int vkey = tid >> 3;              // 0..31
  const int vd0  = (tid & 7) * 16;        // 0..112

  // prologue: tile 0 into buffer 0
  {
    const __bf16* vsrc = vv + (rowbase + vkey) * HD + vd0;
    v8bf vr0 = *(const v8bf*)vsrc, vr1 = *(const v8bf*)(vsrc + 8);
    if (wave == 0)
      tdm_load_2d_bf16((unsigned)(uintptr_t)(&ksh[0][0][0]),
                       kk + rowbase * HD, 32, HD, HD);
#pragma unroll
    for (int i = 0; i < 8; ++i) vT[0][vd0 + i][vkey] = vr0[i];
#pragma unroll
    for (int i = 0; i < 8; ++i) vT[0][vd0 + 8 + i][vkey] = vr1[i];
    if (wave == 0) __builtin_amdgcn_s_wait_tensorcnt(0);
  }
  __syncthreads();

  for (int jb = 0; jb < nblk; ++jb) {
    const int  kb0 = jb * 32;
    const int  cur = jb & 1, nxt = cur ^ 1;
    const bool pf  = (jb + 1 < nblk);

    // prefetch next tile: V -> regs now, K via TDM into the other buffer
    v8bf vr0{}, vr1{};
    if (pf) {
      const __bf16* vsrc = vv + (rowbase + kb0 + 32 + vkey) * HD + vd0;
      vr0 = *(const v8bf*)vsrc;
      vr1 = *(const v8bf*)(vsrc + 8);
      if (wave == 0)
        tdm_load_2d_bf16((unsigned)(uintptr_t)(&ksh[nxt][0][0]),
                         kk + (rowbase + kb0 + 32) * HD, 32, HD, HD);
    }

    // S = q @ k^T  (two 16x16 tiles over 32 keys, K=128 in 4 chunks)
    v8f s0 = v8f{}, s1 = v8f{};
#pragma unroll
    for (int c = 0; c < 4; ++c) {
      const v16bf b0 = frag_b(&ksh[cur][0][c * 32], (lane & 15), HD);
      s0 = wmma_bf16(qf[c], b0, s0);
      const v16bf b1 = frag_b(&ksh[cur][0][c * 32], 16 + (lane & 15), HD);
      s1 = wmma_bf16(qf[c], b1, s1);
    }

    // scale + causal mask + online softmax (row reductions in 16-lane halves)
    const int key0 = kb0 + (lane & 15);
    const int key1 = key0 + 16;
#pragma unroll
    for (int r = 0; r < 8; ++r) {
      const int row = m0 + wave * 16 + r + 8 * hi;
      float a0 = (key0 <= row) ? s0[r] * scale : -__builtin_inff();
      float a1 = (key1 <= row) ? s1[r] * scale : -__builtin_inff();
      float t = fmaxf(a0, a1);
#pragma unroll
      for (int m = 1; m < 16; m <<= 1) t = fmaxf(t, __shfl_xor(t, m, 32));
      const float mn = fmaxf(mst[r], t);
      const float f  = __expf(mst[r] - mn);
      const float p0 = __expf(a0 - mn);
      const float p1 = __expf(a1 - mn);
      float rs = p0 + p1;
#pragma unroll
      for (int m = 1; m < 16; m <<= 1) rs += __shfl_xor(rs, m, 32);
      lst[r] = lst[r] * f + rs;
      mst[r] = mn;
#pragma unroll
      for (int j = 0; j < 8; ++j) oacc[j][r] *= f;
      const int rit = r + 8 * hi;
      psh[wave][rit][lane & 15]        = (__bf16)p0;
      psh[wave][rit][16 + (lane & 15)] = (__bf16)p1;
    }

    // O += P @ V   (A = P 16x32 via LDS relayout, B = V^T tiles)
    const v16bf pa = frag_a(&psh[wave][0][0], (lane & 15), 32);
#pragma unroll
    for (int j = 0; j < 8; ++j) {
      const v16bf bv = frag_b(&vT[cur][j * 16][0], (lane & 15), 32);
      oacc[j] = wmma_bf16(pa, bv, oacc[j]);
    }

    if (pf) {
#pragma unroll
      for (int i = 0; i < 8; ++i) vT[nxt][vd0 + i][vkey] = vr0[i];
#pragma unroll
      for (int i = 0; i < 8; ++i) vT[nxt][vd0 + 8 + i][vkey] = vr1[i];
      if (wave == 0) __builtin_amdgcn_s_wait_tensorcnt(0);
    }
    __syncthreads();
  }

  // normalize and store fp32
#pragma unroll
  for (int r = 0; r < 8; ++r) {
    const float inv = 1.0f / lst[r];
    const int row = m0 + wave * 16 + r + 8 * hi;
#pragma unroll
    for (int j = 0; j < 8; ++j)
      outp[(rowbase + row) * HD + j * 16 + (lane & 15)] = oacc[j][r] * inv;
  }
}

// ---------------- launch ----------------

extern "C" void kernel_launch(void* const* d_in, const int* in_sizes, int n_in,
                              void* d_out, int out_size, void* d_ws, size_t ws_size,
                              hipStream_t stream) {
  (void)in_sizes; (void)n_in; (void)out_size; (void)ws_size;
  const float* x  = (const float*)d_in[0];
  const float* Wq = (const float*)d_in[1];
  const float* Wk = (const float*)d_in[2];
  const float* Wv = (const float*)d_in[3];
  unsigned short* qkv = (unsigned short*)d_ws;   // 3 * 16384*128 bf16 = 12 MB

  qkv_proj_kernel<<<dim3(NROWS / 128, 3), 256, 0, stream>>>(x, Wq, Wk, Wv, qkv);
  attn_kernel<<<dim3(TLEN / 128, BATCH), 256, 0, stream>>>(qkv, (float*)d_out);
}
